// EntropyComponent_76828374991504
// MI455X (gfx1250) — matrix operand order repaired and checked by reading, not compiled, not run on hardware
//
#include <hip/hip_runtime.h>

// ---------------------------------------------------------------------------
// Types for CDNA5 WMMA (wave32)
// ---------------------------------------------------------------------------
typedef __attribute__((ext_vector_type(16))) _Float16 v16h;
typedef __attribute__((ext_vector_type(8)))  _Float16 v8h;
typedef __attribute__((ext_vector_type(8)))  float    v8f;
typedef __attribute__((ext_vector_type(4)))  unsigned int v4u;
typedef __attribute__((ext_vector_type(8)))  int      v8i;
typedef __attribute__((ext_vector_type(4)))  int      v4i;

#define DIN       1024
#define DSTATE    64
#define NHEADS    32      // mamba heads
#define HDIM      32
#define CONV_DIM  1152
#define DPROJ     2208
#define HID       512
#define LSEQ      4096
#define LT        2048
#define BB        2

#if __has_builtin(__builtin_amdgcn_tensor_load_to_lds) && __has_builtin(__builtin_amdgcn_s_wait_tensorcnt)
#define HAVE_TDM 1
#else
#define HAVE_TDM 0
#endif

__device__ __forceinline__ v8h cvt8(float4 a, float4 b) {
    v8h r;
    r[0] = (_Float16)a.x; r[1] = (_Float16)a.y; r[2] = (_Float16)a.z; r[3] = (_Float16)a.w;
    r[4] = (_Float16)b.x; r[5] = (_Float16)b.y; r[6] = (_Float16)b.z; r[7] = (_Float16)b.w;
    return r;
}

__device__ __forceinline__ float block_reduce_sum(float v) {
    #pragma unroll
    for (int off = 16; off > 0; off >>= 1) v += __shfl_xor(v, off, 32);
    __shared__ float red[9];
    const int lane = threadIdx.x & 31, w = threadIdx.x >> 5;
    __syncthreads();
    if (lane == 0) red[w] = v;
    __syncthreads();
    if (threadIdx.x == 0) {
        float s = 0.f;
        const int nw = blockDim.x >> 5;
        for (int i = 0; i < nw; ++i) s += red[i];
        red[8] = s;
    }
    __syncthreads();
    return red[8];
}

#if HAVE_TDM
// ---------------------------------------------------------------------------
// TDM: DMA a 128-row x 32-f32 tile (row stride = k_stride elems, row-major)
// into LDS with 16B padding after each 128B row (LDS row stride 144B).
// D# built per CDNA5 ISA ch.8: group0 = {count|flags, lds_addr, gaddr_lo,
// gaddr_hi|type=2}; group1 = {data_size=4B + pad cfg, dims/strides}.
// ---------------------------------------------------------------------------
__device__ __forceinline__ void tdm_load_tile_a(unsigned lds_byte_addr, const float* gsrc,
                                                unsigned k_stride)
{
    unsigned long long ga = (unsigned long long)(size_t)gsrc;
    v4u g0;
    g0[0] = 1u;                                               // count=1 (valid), user mode
    g0[1] = lds_byte_addr;                                    // lds_addr[31:0]
    g0[2] = (unsigned)ga;                                     // global_addr[31:0]
    g0[3] = ((unsigned)(ga >> 32) & 0x01FFFFFFu) | (2u << 30); // addr[56:32] | type=2
    v8i g1;
    g1[0] = (int)((2u << 16) | (1u << 20) | (4u << 22) | (3u << 25));
    //            data_size=4B  pad_en      per 32 DW    pad 4 DW
    g1[1] = (int)(32u << 16);        // tensor_dim0 = 32 (bits [63:48])
    g1[2] = (int)(128u << 16);       // tensor_dim0 hi=0 | tensor_dim1 lo = 128
    g1[3] = (int)(32u << 16);        // tensor_dim1 hi=0 | tile_dim0 = 32
    g1[4] = (int)128u;               // tile_dim1 = 128, tile_dim2 = 0
    g1[5] = (int)k_stride;           // tensor_dim0_stride lo32 (elements)
    g1[6] = 0;                       // stride hi16 = 0, dim1_stride lo = 0
    g1[7] = 0;
    v4i gz = {};
#if __clang_major__ >= 23
    v8i gz8 = {};
    __builtin_amdgcn_tensor_load_to_lds(g0, g1, gz, gz, gz8, 0);
#else
    __builtin_amdgcn_tensor_load_to_lds(g0, g1, gz, gz, 0);
#endif
}
#endif

// ---------------------------------------------------------------------------
// WMMA GEMM: C[M,N] = A[M,K] @ B[K,N] (+bias) (+res), f32 in/out, f16 WMMA.
// Tile 128x128x32, 8 waves; A staged via TDM (double-buffered f32 in LDS),
// B staged via VALU with f32->f16 convert. Requires M%128==0, K%32==0,
// N%32==0 (N guarded against the 128 tile).
// ---------------------------------------------------------------------------
__launch_bounds__(256)
__global__ void gemm_wmma_kernel(const float* __restrict__ A, const float* __restrict__ Bm,
                                 const float* __restrict__ bias, const float* __restrict__ res,
                                 float* __restrict__ C, int M, int N, int K)
{
    __shared__ __align__(16) float     sAf[2][128][36];   // 36 f32 = 144B row stride
    __shared__ __align__(16) _Float16  sB[32][136];       // 272B row stride
    const int m0 = blockIdx.y << 7;
    const int n0 = blockIdx.x << 7;
    const int t = threadIdx.x;
    const int lane = t & 31, w = t >> 5;
    const int half = lane >> 4, l16 = lane & 15;
    const int brr = t >> 3, bcc = (t & 7) << 4;           // B stage: 32 rows x 128 cols
    v8f acc[8] = {};
    const float* brow = Bm + (size_t)brr * N + n0 + bcc;
    int buf = 0;
#if HAVE_TDM
    if (w == 0)
        tdm_load_tile_a((unsigned)(size_t)&sAf[0][0][0], A + (size_t)m0 * K, (unsigned)K);
#else
    const int arr = t >> 1, ac4 = (t & 1) << 4;
    const float* asrc = A + (size_t)(m0 + arr) * K + ac4;
#endif
    for (int kc = 0; kc < K; kc += 32) {
        const bool have_next = (kc + 32 < K);
#if HAVE_TDM
        if (w == 0 && have_next)
            tdm_load_tile_a((unsigned)(size_t)&sAf[buf ^ 1][0][0],
                            A + (size_t)m0 * K + kc + 32, (unsigned)K);
#else
        {   // manual A stage (fallback)
            const float4* ap = (const float4*)(asrc + kc);
            float4 a0 = ap[0], a1 = ap[1], a2 = ap[2], a3 = ap[3];
            float* dst = &sAf[buf][arr][ac4];
            ((float4*)dst)[0] = a0; ((float4*)dst)[1] = a1;
            ((float4*)dst)[2] = a2; ((float4*)dst)[3] = a3;
        }
#endif
        // stage B 32x128 (guarded against N edge; N%32==0 so v8h never straddles)
        if (have_next) __builtin_prefetch(brow + (size_t)(kc + 32) * N, 0, 3);
        #pragma unroll
        for (int q = 0; q < 2; ++q) {
            v8h hv = {};
            if (n0 + bcc + q * 8 < N) {
                const float* p = brow + (size_t)kc * N + q * 8;
                float4 f0 = ((const float4*)p)[0];
                float4 f1 = ((const float4*)p)[1];
                hv = cvt8(f0, f1);
            }
            *(v8h*)&sB[brr][bcc + q * 8] = hv;
        }
#if HAVE_TDM
        if (w == 0) {
            if (have_next) __builtin_amdgcn_s_wait_tensorcnt(1);
            else           __builtin_amdgcn_s_wait_tensorcnt(0);
        }
#endif
        __syncthreads();
        // A fragment: row = lane%16 of this wave's strip, f32->f16 from LDS
        v16h a;
        {
            const float* arl = &sAf[buf][(w << 4) + l16][0];
            const int kb = half << 3;
            float4 f0 = *(const float4*)(arl + kb);
            float4 f1 = *(const float4*)(arl + kb + 4);
            float4 f2 = *(const float4*)(arl + 16 + kb);
            float4 f3 = *(const float4*)(arl + 20 + kb);
            v8h lo = cvt8(f0, f1), hi = cvt8(f2, f3);
            #pragma unroll
            for (int i = 0; i < 8; ++i) { a[i] = lo[i]; a[i + 8] = hi[i]; }
        }
        // 8 column tiles: b[lane][e] = B[kc+k(half,e)][n0 + jt*16 + lane%16] (table k order)
        #pragma unroll
        for (int jt = 0; jt < 8; ++jt) {
            v16h bf;
            #pragma unroll
            for (int i = 0; i < 8; ++i) { bf[i] = sB[(half << 3) + i][(jt << 4) + l16]; bf[i + 8] = sB[16 + (half << 3) + i][(jt << 4) + l16]; }
            acc[jt] = __builtin_amdgcn_wmma_f32_16x16x32_f16(false, a, false, bf,
                                                             (short)0, acc[jt], false, false);
        }
        __syncthreads();
        buf ^= 1;
    }
    const int mbase = m0 + (w << 4) + (half << 3);
    #pragma unroll
    for (int jt = 0; jt < 8; ++jt) {
        const int col = n0 + (jt << 4) + l16;
        if (col < N) {
            const float bb = bias ? bias[col] : 0.f;
            #pragma unroll
            for (int v = 0; v < 8; ++v) {
                const size_t idx = (size_t)(mbase + v) * N + col;
                float r = acc[jt][v] + bb;
                if (res) r += res[idx];
                C[idx] = r;
            }
        }
    }
}

// ---------------------------------------------------------------------------
// Strided downsample conv (k=3, stride=2, pad (1,1)) as 3 accumulated WMMA
// GEMMs. x:[BB,Lin,512] -> out:[BB,Lin/2,512], wgt:[512][512][3].
// ---------------------------------------------------------------------------
__launch_bounds__(256)
__global__ void dsconv_wmma_kernel(const float* __restrict__ x, const float* __restrict__ wgt,
                                   const float* __restrict__ bias, float* __restrict__ out, int Lin)
{
    __shared__ __align__(16) _Float16 sA[128][48];
    __shared__ __align__(16) _Float16 sB[32][48];
    const int Lout = Lin >> 1;
    const int m0 = blockIdx.y << 7, n0 = blockIdx.x << 5;
    const int t = threadIdx.x, lane = t & 31, w = t >> 5;
    const int half = lane >> 4, l16 = lane & 15;
    const int ar = t >> 1, ac = (t & 1) << 4;
    const int br = t >> 3, bc = (t & 7) << 2;
    const int gr = m0 + ar;
    const int bb = gr / Lout;
    const int lo = gr - bb * Lout;
    v8f acc0 = {}, acc1 = {};
    for (int j = 0; j < 3; ++j) {
        const int il = 2 * lo - 1 + j;
        const bool ok = (il >= 0) && (il < Lin);
        const float* arow = x + ((size_t)bb * Lin + (ok ? il : 0)) * HID + ac;
        for (int kc = 0; kc < HID; kc += 32) {
            v8h h0 = {}, h1 = {};
            if (ok) {
                const float4* ap = (const float4*)(arow + kc);
                float4 a0 = ap[0], a1 = ap[1], a2 = ap[2], a3 = ap[3];
                h0 = cvt8(a0, a1); h1 = cvt8(a2, a3);
            }
            *(v8h*)&sA[ar][ac]     = h0;
            *(v8h*)&sA[ar][ac + 8] = h1;
            #pragma unroll
            for (int cc = 0; cc < 4; ++cc)
                sB[br][bc + cc] = (_Float16)wgt[(size_t)(n0 + bc + cc) * 1536 + (kc + br) * 3 + j];
            __syncthreads();
            const int row = (w << 4) + l16;
            const int kb = half << 3;
            v8h alo = *(const v8h*)&sA[row][kb];
            v8h ahi = *(const v8h*)&sA[row][16 + kb];
            v16h a;
            #pragma unroll
            for (int i = 0; i < 8; ++i) { a[i] = alo[i]; a[i + 8] = ahi[i]; }
            v16h b0, b1;
            #pragma unroll
            for (int i = 0; i < 8; ++i) { b0[i] = sB[kb + i][l16]; b0[i + 8] = sB[16 + kb + i][l16];
                                          b1[i] = sB[kb + i][16 + l16]; b1[i + 8] = sB[16 + kb + i][16 + l16]; }
            acc0 = __builtin_amdgcn_wmma_f32_16x16x32_f16(false, a, false, b0, (short)0, acc0, false, false);
            acc1 = __builtin_amdgcn_wmma_f32_16x16x32_f16(false, a, false, b1, (short)0, acc1, false, false);
            __syncthreads();
        }
    }
    const int mbase = m0 + (w << 4) + (half << 3);
    const int c0 = n0 + l16, c1 = c0 + 16;
    #pragma unroll
    for (int v = 0; v < 8; ++v) {
        const size_t i0 = (size_t)(mbase + v) * HID + c0;
        const size_t i1 = (size_t)(mbase + v) * HID + c1;
        out[i0] = acc0[v] + bias[c0];
        out[i1] = acc1[v] + bias[c1];
    }
}

// ---------------------------------------------------------------------------
// Flash attention (non-causal), 8 heads, hd=64. 4 waves x 16 query rows per
// block. qkv:[BB,LT,1536] (q|k|v), out:[BB,LT,512].
// ---------------------------------------------------------------------------
__launch_bounds__(128)
__global__ void attn_wmma_kernel(const float* __restrict__ qkv, float* __restrict__ out, int Lq)
{
    __shared__ __align__(16) _Float16 sK[64][56];        // [dim][token]
    __shared__ __align__(16) _Float16 sV[32][72];        // [token][dim]
    __shared__ __align__(16) _Float16 sP[4][16][48];     // per-wave P tile
    const int b = blockIdx.z, h = blockIdx.y;
    const int q0 = blockIdx.x << 6;
    const int t = threadIdx.x, lane = t & 31, w = t >> 5;
    const int half = lane >> 4, l16 = lane & 15;
    v16h aqv[2];
    {
        const int qrow = q0 + (w << 4) + l16;
        const float* qp = qkv + ((size_t)b * Lq + qrow) * 1536 + h * 64;
        #pragma unroll
        for (int c = 0; c < 2; ++c) {
            #pragma unroll
            for (int g = 0; g < 2; ++g) {
                const float* p = qp + c * 32 + g * 16 + (half << 3);
                float4 f0 = ((const float4*)p)[0];
                float4 f1 = ((const float4*)p)[1];
                aqv[c][g * 8 + 0] = (_Float16)(f0.x * 0.125f);
                aqv[c][g * 8 + 1] = (_Float16)(f0.y * 0.125f);
                aqv[c][g * 8 + 2] = (_Float16)(f0.z * 0.125f);
                aqv[c][g * 8 + 3] = (_Float16)(f0.w * 0.125f);
                aqv[c][g * 8 + 4] = (_Float16)(f1.x * 0.125f);
                aqv[c][g * 8 + 5] = (_Float16)(f1.y * 0.125f);
                aqv[c][g * 8 + 6] = (_Float16)(f1.z * 0.125f);
                aqv[c][g * 8 + 7] = (_Float16)(f1.w * 0.125f);
            }
        }
    }
    v8f o[4] = {};
    float mrow[8], lrow[8];
    #pragma unroll
    for (int v = 0; v < 8; ++v) { mrow[v] = -1e30f; lrow[v] = 0.f; }

    for (int kc = 0; kc < Lq; kc += 32) {
        __syncthreads();
        {   // stage K (transposed) and V for 32 tokens
            const int tok = t >> 2, dbase = (t & 3) << 4;
            const float* kp = qkv + ((size_t)b * Lq + kc + tok) * 1536 + 512 + h * 64 + dbase;
            const float* vp = kp + 512;
            #pragma unroll
            for (int i = 0; i < 16; ++i) sK[dbase + i][tok] = (_Float16)kp[i];
            #pragma unroll
            for (int i = 0; i < 16; ++i) sV[tok][dbase + i] = (_Float16)vp[i];
        }
        __syncthreads();
        v8f s0 = {}, s1 = {};
        #pragma unroll
        for (int c = 0; c < 2; ++c) {
            const v16h aq = c ? aqv[1] : aqv[0];
            const int kd = c * 32 + (half << 3);
            v16h bk0, bk1;
            #pragma unroll
            for (int i = 0; i < 8; ++i) { bk0[i] = sK[kd + i][l16]; bk0[i + 8] = sK[kd + 16 + i][l16];
                                          bk1[i] = sK[kd + i][16 + l16]; bk1[i + 8] = sK[kd + 16 + i][16 + l16]; }
            s0 = __builtin_amdgcn_wmma_f32_16x16x32_f16(false, aq, false, bk0, (short)0, s0, false, false);
            s1 = __builtin_amdgcn_wmma_f32_16x16x32_f16(false, aq, false, bk1, (short)0, s1, false, false);
        }
        #pragma unroll
        for (int v = 0; v < 8; ++v) {
            float mx = fmaxf(s0[v], s1[v]);
            mx = fmaxf(mx, __shfl_xor(mx, 1, 32));
            mx = fmaxf(mx, __shfl_xor(mx, 2, 32));
            mx = fmaxf(mx, __shfl_xor(mx, 4, 32));
            mx = fmaxf(mx, __shfl_xor(mx, 8, 32));
            const float mnew = fmaxf(mrow[v], mx);
            const float corr = __expf(mrow[v] - mnew);
            mrow[v] = mnew;
            const float p0 = __expf(s0[v] - mnew);
            const float p1 = __expf(s1[v] - mnew);
            float rs = p0 + p1;
            rs += __shfl_xor(rs, 1, 32);
            rs += __shfl_xor(rs, 2, 32);
            rs += __shfl_xor(rs, 4, 32);
            rs += __shfl_xor(rs, 8, 32);
            lrow[v] = lrow[v] * corr + rs;
            o[0][v] *= corr; o[1][v] *= corr; o[2][v] *= corr; o[3][v] *= corr;
            sP[w][(half << 3) + v][l16]      = (_Float16)p0;
            sP[w][(half << 3) + v][16 + l16] = (_Float16)p1;
        }
        __syncthreads();
        v16h apv;
        {
            const _Float16* pr = &sP[w][l16][half << 3];
            v8h plo = *(const v8h*)pr;
            v8h phi = *(const v8h*)(pr + 16);
            #pragma unroll
            for (int i = 0; i < 8; ++i) { apv[i] = plo[i]; apv[i + 8] = phi[i]; }
        }
        #pragma unroll
        for (int jt = 0; jt < 4; ++jt) {
            v16h bv;
            #pragma unroll
            for (int i = 0; i < 8; ++i) { bv[i] = sV[(half << 3) + i][(jt << 4) + l16]; bv[i + 8] = sV[16 + (half << 3) + i][(jt << 4) + l16]; }
            o[jt] = __builtin_amdgcn_wmma_f32_16x16x32_f16(false, apv, false, bv, (short)0, o[jt], false, false);
        }
    }
    #pragma unroll
    for (int v = 0; v < 8; ++v) {
        const int row = q0 + (w << 4) + (half << 3) + v;
        const float inv = 1.f / lrow[v];
        float* op = out + ((size_t)b * Lq + row) * HID + h * 64 + l16;
        op[0]  = o[0][v] * inv;
        op[16] = o[1][v] * inv;
        op[32] = o[2][v] * inv;
        op[48] = o[3][v] * inv;
    }
}

// ---------------------------------------------------------------------------
// Depthwise causal conv (k=4) + bias + SiLU over the xBC slice of zxbcdt.
// ---------------------------------------------------------------------------
__launch_bounds__(256)
__global__ void dwconv_silu_kernel(const float* __restrict__ zx, const float* __restrict__ cw,
                                   const float* __restrict__ cb, float* __restrict__ out,
                                   int total, int L)
{
    const int idx = blockIdx.x * 256 + threadIdx.x;
    if (idx >= total) return;
    const int c = idx % CONV_DIM;
    const int bl = idx / CONV_DIM;
    const int l = bl % L;
    const float* p = zx + (size_t)bl * DPROJ + DIN + c;
    float acc = cb[c];
    #pragma unroll
    for (int j = 0; j < 4; ++j) {
        const int ls = l - 3 + j;
        if (ls >= 0) acc = fmaf(p[(ls - l) * DPROJ], cw[c * 4 + j], acc);
    }
    acc = acc / (1.f + __expf(-acc));
    out[(size_t)bl * CONV_DIM + c] = acc;
}

// ---------------------------------------------------------------------------
// Selective-scan recurrence. One block per (head, batch); state (32x64) f32
// in registers, 8 elements/lane, y reduced over the 8-lane n-groups.
// ---------------------------------------------------------------------------
__launch_bounds__(256)
__global__ void mamba_scan_kernel(const float* __restrict__ zx, const float* __restrict__ xc,
                                  const float* __restrict__ dtb, const float* __restrict__ Alog,
                                  const float* __restrict__ Dp, float* __restrict__ y, int L)
{
    const int hh = blockIdx.x, b = blockIdx.y;
    const int t = threadIdx.x;
    const int p = t >> 3, ng = (t & 7) << 3;
    const float A = __expf(Alog[hh]);
    const float dtbias = dtb[hh];
    const float Dv = Dp[hh];
    float hreg[8] = {};
    const float* zrow = zx + (size_t)b * L * DPROJ + (DIN + CONV_DIM) + hh;
    const float* xrow = xc + (size_t)b * L * CONV_DIM;
    float* yrow = y + (size_t)b * L * DIN + hh * HDIM + p;
    for (int l = 0; l < L; ++l) {
        float dtv = zrow[(size_t)l * DPROJ] + dtbias;
        dtv = dtv > 20.f ? dtv : log1pf(__expf(dtv));          // softplus
        const float dec = __expf(-A * dtv);
        const float* r = xrow + (size_t)l * CONV_DIM;
        const float xp = r[hh * HDIM + p];
        const float dx = dtv * xp;
        float4 B0 = *(const float4*)(r + DIN + ng);
        float4 B1 = *(const float4*)(r + DIN + ng + 4);
        float4 C0 = *(const float4*)(r + DIN + DSTATE + ng);
        float4 C1 = *(const float4*)(r + DIN + DSTATE + ng + 4);
        const float Bv[8] = {B0.x, B0.y, B0.z, B0.w, B1.x, B1.y, B1.z, B1.w};
        const float Cv[8] = {C0.x, C0.y, C0.z, C0.w, C1.x, C1.y, C1.z, C1.w};
        float yacc = 0.f;
        #pragma unroll
        for (int i = 0; i < 8; ++i) {
            hreg[i] = fmaf(hreg[i], dec, dx * Bv[i]);
            yacc = fmaf(hreg[i], Cv[i], yacc);
        }
        yacc += __shfl_xor(yacc, 1, 32);
        yacc += __shfl_xor(yacc, 2, 32);
        yacc += __shfl_xor(yacc, 4, 32);
        if ((t & 7) == 0) yrow[(size_t)l * DIN] = yacc + Dv * xp;
    }
}

// ---------------------------------------------------------------------------
// out = rmsnorm(y * silu(z), nw); z is the first DIN cols of zxbcdt.
// ---------------------------------------------------------------------------
__launch_bounds__(256)
__global__ void gate_rms_kernel(const float* __restrict__ y, const float* __restrict__ zx,
                                const float* __restrict__ nw, float* __restrict__ out)
{
    const size_t r = blockIdx.x;
    const float* yr = y + r * DIN;
    const float* zr = zx + r * DPROJ;
    float tv[4]; float ss = 0.f;
    #pragma unroll
    for (int i = 0; i < 4; ++i) {
        const int c = threadIdx.x + (i << 8);
        const float z = zr[c];
        const float g = yr[c] * (z / (1.f + __expf(-z)));
        tv[i] = g; ss += g * g;
    }
    const float tot = block_reduce_sum(ss);
    const float inv = rsqrtf(tot * (1.f / DIN) + 1e-5f);
    #pragma unroll
    for (int i = 0; i < 4; ++i) {
        const int c = threadIdx.x + (i << 8);
        out[r * DIN + c] = tv[i] * inv * nw[c];
    }
}

__launch_bounds__(256)
__global__ void rms512_kernel(const float* __restrict__ in, const float* __restrict__ wgt,
                              float* __restrict__ out)
{
    const size_t r = blockIdx.x;
    const int c0 = threadIdx.x, c1 = threadIdx.x + 256;
    const float x0 = in[r * HID + c0], x1 = in[r * HID + c1];
    const float tot = block_reduce_sum(x0 * x0 + x1 * x1);
    const float inv = rsqrtf(tot * (1.f / HID) + 1e-5f);
    out[r * HID + c0] = x0 * inv * wgt[c0];
    out[r * HID + c1] = x1 * inv * wgt[c1];
}

__launch_bounds__(256)
__global__ void ln512_kernel(const float* __restrict__ in, const float* __restrict__ wgt,
                             const float* __restrict__ bs, float* __restrict__ out)
{
    const size_t r = blockIdx.x;
    const int c0 = threadIdx.x, c1 = threadIdx.x + 256;
    const float x0 = in[r * HID + c0], x1 = in[r * HID + c1];
    const float m = block_reduce_sum(x0 + x1) * (1.f / HID);
    const float d0 = x0 - m, d1 = x1 - m;
    const float var = block_reduce_sum(d0 * d0 + d1 * d1) * (1.f / HID);
    const float inv = rsqrtf(var + 1e-5f);
    out[r * HID + c0] = d0 * inv * wgt[c0] + bs[c0];
    out[r * HID + c1] = d1 * inv * wgt[c1] + bs[c1];
}

__launch_bounds__(256)
__global__ void gelu_kernel(float* __restrict__ x, int n)
{
    const int i = blockIdx.x * 256 + threadIdx.x;
    if (i < n) {
        const float v = x[i];
        x[i] = 0.5f * v * (1.f + erff(v * 0.70710678118f));
    }
}

// ---------------------------------------------------------------------------
// Host side
// ---------------------------------------------------------------------------
static void launch_gemm(hipStream_t s, const float* A, const float* B, const float* bias,
                        const float* res, float* C, int M, int N, int K)
{
    gemm_wmma_kernel<<<dim3((N + 127) / 128, M / 128), 256, 0, s>>>(A, B, bias, res, C, M, N, K);
}

static void run_mamba(hipStream_t s, const float* xin,
                      const float* Wi, const float* cw, const float* cb, const float* dtbp,
                      const float* Alog, const float* Dp, const float* nw, const float* Wo,
                      const float* nouter,
                      float* ZX, float* XC, float* YB, float* YG, float* xout)
{
    const int M = BB * LSEQ;                 // 8192
    launch_gemm(s, xin, Wi, nullptr, nullptr, ZX, M, DPROJ, HID);
    const int total = M * CONV_DIM;
    dwconv_silu_kernel<<<(total + 255) / 256, 256, 0, s>>>(ZX, cw, cb, XC, total, LSEQ);
    mamba_scan_kernel<<<dim3(NHEADS, BB), 256, 0, s>>>(ZX, XC, dtbp, Alog, Dp, YB, LSEQ);
    gate_rms_kernel<<<M, 256, 0, s>>>(YB, ZX, nw, YG);
    // MO aliases YB (scan output is dead after gate_rms)
    launch_gemm(s, YG, Wo, nullptr, xin, YB, M, HID, DIN);
    rms512_kernel<<<M, 256, 0, s>>>(YB, nouter, xout);
}

extern "C" void kernel_launch(void* const* d_in, const int* in_sizes, int n_in,
                              void* d_out, int out_size, void* d_ws, size_t ws_size,
                              hipStream_t stream)
{
    (void)in_sizes; (void)n_in; (void)out_size; (void)ws_size;
    const float** in = (const float**)d_in;
    const float* x    = in[0];
    const float* Wp   = in[1];
    const float* bp   = in[2];
    const float* n1w  = in[19];
    const float* n2w  = in[20];
    const float* dsw  = in[21];
    const float* dsb  = in[22];
    const float* Wqkv = in[23];
    const float* bqkv = in[24];
    const float* tWo  = in[25];
    const float* tbo  = in[26];
    const float* W1   = in[27];
    const float* b1   = in[28];
    const float* W2   = in[29];
    const float* b2   = in[30];
    const float* ln1w = in[31];
    const float* ln1b = in[32];
    const float* ln2w = in[33];
    const float* ln2b = in[34];
    const float* onw  = in[35];
    const float* onb  = in[36];

    const int M  = BB * LSEQ;   // 8192
    const int Mh = BB * LT;     // 4096

    // Workspace layout (floats); peak ~211 MB
    float* ws = (float*)d_ws;
    float* X0 = ws;                                   // [M,512]
    float* X1 = X0 + (size_t)M * HID;                 // [M,512]
    float* ZX = X1 + (size_t)M * HID;                 // [M,2208]
    float* XC = ZX + (size_t)M * DPROJ;               // [M,1152]
    float* YB = XC + (size_t)M * CONV_DIM;            // [M,1024]
    float* YG = YB + (size_t)M * DIN;                 // [M,1024]
    // Transformer-stage aliases (mamba buffers are dead by then)
    float* XT  = XC;                                  // [Mh,512] downsampled x
    float* QKV = ZX;                                  // [Mh,1536]
    float* AO  = YB;                                  // [Mh,512]
    float* T1  = YG;                                  // [Mh,512]
    float* T1n = YG + (size_t)Mh * HID;               // [Mh,512]
    float* T2  = YG + (size_t)2 * Mh * HID;           // [Mh,512]
    float* FF  = X1;                                  // [Mh,1024]
    float* T2n = YB;                                  // [Mh,512] (AO dead)

    // 1) input projection
    launch_gemm(stream, x, Wp, bp, nullptr, X0, M, HID, 1024);

    // 2) mamba block 1 (+res, rmsnorm n1)
    run_mamba(stream, X0, in[3], in[4], in[5], in[6], in[7], in[8], in[9], in[10],
              n1w, ZX, XC, YB, YG, X1);
    // 3) mamba block 2 (+res, rmsnorm n2)
    run_mamba(stream, X1, in[11], in[12], in[13], in[14], in[15], in[16], in[17], in[18],
              n2w, ZX, XC, YB, YG, X0);

    // 4) strided downsample conv (k=3, s=2)
    dsconv_wmma_kernel<<<dim3(HID / 32, Mh / 128), 256, 0, stream>>>(X0, dsw, dsb, XT, LSEQ);

    // 5) transformer layer
    launch_gemm(stream, XT, Wqkv, bqkv, nullptr, QKV, Mh, 3 * HID, HID);
    attn_wmma_kernel<<<dim3(LT / 64, 8, BB), 128, 0, stream>>>(QKV, AO, LT);
    launch_gemm(stream, AO, tWo, tbo, XT, T1, Mh, HID, HID);
    ln512_kernel<<<Mh, 256, 0, stream>>>(T1, ln1w, ln1b, T1n);
    launch_gemm(stream, T1n, W1, b1, nullptr, FF, Mh, 1024, HID);
    gelu_kernel<<<(Mh * 1024 + 255) / 256, 256, 0, stream>>>(FF, Mh * 1024);
    launch_gemm(stream, FF, W2, b2, T1n, T2, Mh, HID, 1024);
    ln512_kernel<<<Mh, 256, 0, stream>>>(T2, ln2w, ln2b, T2n);

    // 6) final layernorm -> output
    ln512_kernel<<<Mh, 256, 0, stream>>>(T2n, onw, onb, (float*)d_out);
}
